// CompressedKVModelWrapper_17669495455889
// MI455X (gfx1250) — matrix operand order repaired and checked by reading, not compiled, not run on hardware
//
#include <hip/hip_runtime.h>
#include <hip/hip_bf16.h>
#include <stdint.h>

// ---------------------------------------------------------------------------
// CompressedKV attention for MI455X (gfx1250, wave32, WMMA f16 16x16x32 + TDM).
// Pipeline:
//   0) cvt hidden->f16, transpose+cvt w_attn/w_proj -> f16 [N][K]
//   1) qkv = X @ Wattn + b   (WMMA GEMM, double-buffered TENSOR_LOAD_TO_LDS)
//   2) W_keffT[h] = (wk_c[h] @ wk_d[h])^T  (exact linear folding)
//   3) k_dec = k @ W_keff (row-major), v_decT = (v @ W_veff)^T  (WMMA per head)
//   4) causal flash attention (WMMA QK^T and P@V, online softmax)
//   5) out = attn @ Wproj + b  (WMMA GEMM via TDM, f32 output)
// ---------------------------------------------------------------------------

typedef __attribute__((ext_vector_type(16))) _Float16     v16h;
typedef __attribute__((ext_vector_type(8)))  float        v8f;
typedef __attribute__((ext_vector_type(4)))  unsigned int u32x4;
typedef __attribute__((ext_vector_type(8)))  int          i32x8;
typedef __attribute__((ext_vector_type(4)))  int          i32x4;

union V16 { v16h v; unsigned int u[8]; };

__device__ __forceinline__ v8f wmma_f16(const v16h& a, const v16h& b, const v8f& c) {
    return __builtin_amdgcn_wmma_f32_16x16x32_f16(
        false, a, false, b, (short)0, c, false, false);
}

// A-matrix (16x32 f16) packed-u32 slot j -> K base index (contiguous: K = 16*hi + 2j, matches B).
__device__ __forceinline__ int afrag_kbase(int j, int hi) {
    return hi * 16 + 2 * j;
}

#if __has_builtin(__builtin_amdgcn_tensor_load_to_lds)
#define USE_TDM 1
#else
#define USE_TDM 0
#endif

#if USE_TDM
// D# group1 for a [128 x 32] f16 tile, row stride lda (elems).  Loop-invariant.
__device__ __forceinline__ i32x8 tdm_group1_f16(int lda_elems, int tdim0, int tdim1) {
    i32x8 g1;
    g1[0] = (1 << 16);                                     // data_size=1 (2 bytes)
    g1[1] = (int)(((unsigned)tdim0 & 0xffffu) << 16);      // tensor_dim0[15:0]
    g1[2] = (int)(((unsigned)tdim0 >> 16) & 0xffffu)       // tensor_dim0[31:16]
          | (int)(((unsigned)tdim1 & 0xffffu) << 16);      // tensor_dim1[15:0]
    g1[3] = (int)(((unsigned)tdim1 >> 16) & 0xffffu)       // tensor_dim1[31:16]
          | (32 << 16);                                    // tile_dim0 = 32
    g1[4] = 128;                                           // tile_dim1=128, tile_dim2=0
    g1[5] = lda_elems;                                     // tensor_dim0_stride[31:0]
    g1[6] = 0;
    g1[7] = 0;
    return g1;
}

// Per-issue: only the addresses change.
__device__ __forceinline__ void tdm_issue_f16(unsigned lds_byte_addr,
                                              const _Float16* gptr, i32x8 g1) {
    unsigned long long ga = (unsigned long long)(uintptr_t)gptr;
    u32x4 g0;
    g0[0] = 1u;                                            // count=1
    g0[1] = lds_byte_addr;                                 // lds_addr
    g0[2] = (unsigned)(ga & 0xffffffffull);                // global_addr[31:0]
    g0[3] = (unsigned)((ga >> 32) & 0x1ffffffull)          // global_addr[56:32]
          | (2u << 30);                                    // type=2 ("image")
    i32x4 gz = {0, 0, 0, 0};
#if __clang_major__ >= 23
    i32x8 gz8 = {};
    __builtin_amdgcn_tensor_load_to_lds(g0, g1, gz, gz, gz8, 0);
#else
    __builtin_amdgcn_tensor_load_to_lds(g0, g1, gz, gz, 0);
#endif
}
#endif

// ===========================================================================
// Elementwise f32 -> f16
// ===========================================================================
__global__ void cvt_kernel(const float* __restrict__ in, _Float16* __restrict__ out, int n) {
    int i = blockIdx.x * blockDim.x + threadIdx.x;
    if (i < n) out[i] = (_Float16)in[i];
}

// in[R][C] -> out[C][R] with f32->f16 (coalesced writes)
__global__ void transpose_cvt_kernel(const float* __restrict__ in,
                                     _Float16* __restrict__ out, int R, int C) {
    int idx = blockIdx.x * blockDim.x + threadIdx.x;
    if (idx < R * C) {
        int c = idx / R, r = idx - c * R;
        out[idx] = (_Float16)in[(size_t)r * C + c];
    }
}

// ===========================================================================
// WMMA GEMM, all-f16 inputs: C[M,N] = A[M,K] * BT[N,K]^T + bias[N]
// 256 threads (8 waves as 4x2), block tile 128x128, K-step 32, K compile-time.
// Tiles double-buffered in LDS via the Tensor Data Mover: tile i+1's DMA
// overlaps tile i's WMMAs; TENSORcnt<=2 gates the current buffer; the loop is
// unrolled x2 so the buffer select is static.
// ===========================================================================
template <typename OT, int K>
__global__ __launch_bounds__(256)
void wmma_gemm_f16(const _Float16* __restrict__ A,   // [M][K]
                   const _Float16* __restrict__ BT,  // [N][K]
                   const float* __restrict__ bias,
                   OT* __restrict__ Cmat, int M, int N) {
    constexpr int NK = K >> 5;                           // K-steps
    constexpr unsigned TILE_B = 128 * 32 * 2;            // bytes per LDS tile
    __shared__ __align__(16) _Float16 As [2][128 * 32];  // [buf][row][k]
    __shared__ __align__(16) _Float16 BsT[2][128 * 32];  // [buf][col][k]

    const int t    = threadIdx.x;
    const int lane = t & 31;
    const int wave = t >> 5;
    const int waveM = wave >> 1;
    const int waveN = wave & 1;
    const int hi = lane >> 4;
    const int ln = lane & 15;

    const int m0 = blockIdx.y * 128;
    const int n0 = blockIdx.x * 128;

    const _Float16* Abase = A  + (size_t)m0 * K;
    const _Float16* Bbase = BT + (size_t)n0 * K;

    v8f acc[2][4] = {};

    // 8 WMMAs on one staged tile pair.
    auto compute_tile = [&](const _Float16* Asb, const _Float16* Bsb) {
        V16 af[2];
        #pragma unroll
        for (int r = 0; r < 2; ++r) {
            int row = waveM * 32 + r * 16 + ln;
            const unsigned int* p = (const unsigned int*)&Asb[row * 32];
            #pragma unroll
            for (int j = 0; j < 8; ++j)
                af[r].u[j] = p[afrag_kbase(j, hi) >> 1];
        }
        V16 bf[4];
        #pragma unroll
        for (int c = 0; c < 4; ++c) {
            int col = waveN * 64 + c * 16 + ln;
            const unsigned int* p = (const unsigned int*)&Bsb[col * 32];
            #pragma unroll
            for (int j = 0; j < 8; ++j)
                bf[c].u[j] = p[j + 8 * hi];              // K = 2j + 16*hi
        }
        #pragma unroll
        for (int r = 0; r < 2; ++r)
            #pragma unroll
            for (int c = 0; c < 4; ++c)
                acc[r][c] = wmma_f16(af[r].v, bf[c].v, acc[r][c]);
    };

#if USE_TDM
    const unsigned asAddr = (unsigned)(uintptr_t)(void*)&As [0][0];
    const unsigned bsAddr = (unsigned)(uintptr_t)(void*)&BsT[0][0];
    const i32x8 g1d = tdm_group1_f16(K, K, 1 << 20);

    if (t == 0) {                                  // prologue: tile 0 -> buf 0
        tdm_issue_f16(asAddr, Abase, g1d);
        tdm_issue_f16(bsAddr, Bbase, g1d);
    }
    #pragma unroll 2
    for (int i = 0; i < NK - 1; ++i) {
        const int buf = i & 1;
        if (t == 0) {                              // prefetch tile i+1
            unsigned boff = (unsigned)((buf ^ 1) * TILE_B);
            tdm_issue_f16(asAddr + boff, Abase + ((i + 1) << 5), g1d);
            tdm_issue_f16(bsAddr + boff, Bbase + ((i + 1) << 5), g1d);
        }
        __builtin_amdgcn_s_wait_tensorcnt((short)2);   // tile i landed
        __syncthreads();
        compute_tile(&As[buf][0], &BsT[buf][0]);
        __syncthreads();                               // readers done before DMA
    }
    __builtin_amdgcn_s_wait_tensorcnt((short)0);       // last tile landed
    __syncthreads();
    compute_tile(&As[(NK - 1) & 1][0], &BsT[(NK - 1) & 1][0]);
#else
    for (int i = 0; i < NK; ++i) {
        const int buf = i & 1;
        int kk = i << 5;
        #pragma unroll
        for (int x = 0; x < 16; ++x) {
            int idx = t * 16 + x;
            int r = idx >> 5, c = idx & 31;
            As [buf][idx] = Abase[(size_t)r * K + kk + c];
            BsT[buf][idx] = Bbase[(size_t)r * K + kk + c];
        }
        __syncthreads();
        compute_tile(&As[buf][0], &BsT[buf][0]);
        __syncthreads();
    }
#endif

    // ---- epilogue (C layout: VGPR e -> M = e + 8*hi, N = ln) ----
    #pragma unroll
    for (int r = 0; r < 2; ++r) {
        #pragma unroll
        for (int c = 0; c < 4; ++c) {
            int col = n0 + waveN * 64 + c * 16 + ln;
            float bv = bias ? bias[col] : 0.0f;
            #pragma unroll
            for (int e = 0; e < 8; ++e) {
                int row = m0 + waveM * 32 + r * 16 + e + 8 * hi;
                Cmat[(size_t)row * N + col] = (OT)(acc[r][c][e] + bv);
            }
        }
    }
}

// ===========================================================================
// Effective per-head weight, stored TRANSPOSED: outT[h][d2][d] = sum_c
// wc[h][d][c] * wd[h][c][d2].  H=16, hd=64, C=32 -> 65536 outputs.
// ===========================================================================
__global__ void eff_weight_kernel(const float* __restrict__ wc,
                                  const float* __restrict__ wd,
                                  _Float16* __restrict__ outT) {
    int idx = blockIdx.x * blockDim.x + threadIdx.x;   // < 16*64*64
    int h  = idx >> 12;
    int d  = (idx >> 6) & 63;
    int d2 = idx & 63;
    float s = 0.0f;
    #pragma unroll
    for (int c = 0; c < 32; ++c)
        s += wc[(h * 64 + d) * 32 + c] * wd[(h * 32 + c) * 64 + d2];
    outT[((h * 64 + d2) << 6) + d] = (_Float16)s;
}

// ===========================================================================
// k_dec/v_dec: per (b,h): dec = qkv_slice[2048,64] @ Weff[h]
// WeffT is [d_out][d_in].  transposed=0: dec[S][64]; transposed=1: dec[64][S].
// ===========================================================================
__global__ __launch_bounds__(256)
void kv_dec_kernel(const _Float16* __restrict__ qkv,
                   const _Float16* __restrict__ WeffT,
                   _Float16* __restrict__ dec, int col_off, int transposed) {
    const int bh = blockIdx.y;
    const int b = bh >> 4, h = bh & 15;
    const int wave = threadIdx.x >> 5;
    const int lane = threadIdx.x & 31;
    const int hi = lane >> 4, ln = lane & 15;
    const int s0 = blockIdx.x * 128 + wave * 16;

    const _Float16* Arow = qkv + (size_t)(b * 2048) * 3072 + col_off + h * 64;
    const _Float16* Bp   = WeffT + h * 4096;   // [d2][d]

    v8f acc[4] = {};
    #pragma unroll
    for (int ks = 0; ks < 2; ++ks) {          // K = 64 -> two x32 steps
        V16 af;
        {
            const unsigned int* p =
                (const unsigned int*)(Arow + (size_t)(s0 + ln) * 3072 + ks * 32);
            #pragma unroll
            for (int j = 0; j < 8; ++j)
                af.u[j] = p[afrag_kbase(j, hi) >> 1];
        }
        #pragma unroll
        for (int c = 0; c < 4; ++c) {
            V16 bf;   // B: K = d = ks*32 + 2j + 16*hi (contiguous pairs)
            const unsigned int* p =
                (const unsigned int*)(Bp + (c * 16 + ln) * 64);
            #pragma unroll
            for (int j = 0; j < 8; ++j)
                bf.u[j] = p[ks * 16 + j + 8 * hi];
            acc[c] = wmma_f16(af.v, bf.v, acc[c]);
        }
    }
    #pragma unroll
    for (int c = 0; c < 4; ++c)
        #pragma unroll
        for (int e = 0; e < 8; ++e) {
            int row = s0 + e + 8 * hi;
            int col = c * 16 + ln;
            _Float16 val = (_Float16)acc[c][e];
            if (transposed)
                dec[(size_t)bh * 64 * 2048 + (size_t)col * 2048 + row] = val;
            else
                dec[((size_t)bh * 2048 + row) * 64 + col] = val;
        }
}

// ===========================================================================
// Causal flash attention, one wave per (b, h, 16-row query tile).
// kdec is [S][hd] (for QK^T B-frags), vdecT is [hd][S] (for P@V B-frags).
// ===========================================================================
__global__ __launch_bounds__(32)
void flash_attn_kernel(const _Float16* __restrict__ qkv,
                       const _Float16* __restrict__ kdec,
                       const _Float16* __restrict__ vdecT,
                       _Float16* __restrict__ attn) {
    const int s0 = blockIdx.x * 16;
    const int h  = blockIdx.y;
    const int b  = blockIdx.z;
    const int lane = threadIdx.x;
    const int hi = lane >> 4, ln = lane & 15;

    __shared__ __align__(16) _Float16 Pbuf[16 * 32];

    const _Float16* Qp = qkv + (size_t)(b * 2048) * 3072 + h * 64;
    V16 qf[2];
    #pragma unroll
    for (int ks = 0; ks < 2; ++ks) {
        const unsigned int* p =
            (const unsigned int*)(Qp + (size_t)(s0 + ln) * 3072 + ks * 32);
        #pragma unroll
        for (int j = 0; j < 8; ++j)
            qf[ks].u[j] = p[afrag_kbase(j, hi) >> 1];
    }

    const _Float16* Kp  = kdec  + (size_t)(b * 16 + h) * 2048 * 64;  // [S][64]
    const _Float16* VpT = vdecT + (size_t)(b * 16 + h) * 64 * 2048;  // [64][S]

    v8f o[4] = {};
    float m[8], l[8];
    #pragma unroll
    for (int e = 0; e < 8; ++e) { m[e] = -1e30f; l[e] = 0.0f; }

    const int kend = s0 + 16;
    for (int kb = 0; kb < kend; kb += 32) {
        // ---- scores: two 16x16 tiles ----
        v8f s[2] = {};
        #pragma unroll
        for (int half = 0; half < 2; ++half) {
            #pragma unroll
            for (int ks = 0; ks < 2; ++ks) {
                V16 bf;   // B = Kdec^T: N = key (=ln), K = hd = ks*32+2j+16*hi
                const unsigned int* p = (const unsigned int*)
                    (Kp + (size_t)(kb + half * 16 + ln) * 64 + ks * 32);
                #pragma unroll
                for (int j = 0; j < 8; ++j)
                    bf.u[j] = p[j + 8 * hi];
                s[half] = wmma_f16(qf[ks].v, bf.v, s[half]);
            }
        }
        // ---- scale + causal mask + online softmax ----
        #pragma unroll
        for (int e = 0; e < 8; ++e) {
            int qrow = s0 + e + 8 * hi;
            float v0 = s[0][e] * 0.125f;
            float v1 = s[1][e] * 0.125f;
            if (kb + ln > qrow)      v0 = -1e9f;
            if (kb + 16 + ln > qrow) v1 = -1e9f;
            s[0][e] = v0; s[1][e] = v1;
        }
        float psum[8];
        #pragma unroll
        for (int e = 0; e < 8; ++e) {
            float mx = fmaxf(s[0][e], s[1][e]);
            #pragma unroll
            for (int off = 1; off < 16; off <<= 1)
                mx = fmaxf(mx, __shfl_xor(mx, off, 32));
            float mnew = fmaxf(m[e], mx);
            float a = __expf(m[e] - mnew);
            m[e] = mnew;
            l[e] *= a;
            #pragma unroll
            for (int c = 0; c < 4; ++c) o[c][e] *= a;
            float p0 = __expf(s[0][e] - mnew);
            float p1 = __expf(s[1][e] - mnew);
            Pbuf[(e + 8 * hi) * 32 + ln]      = (_Float16)p0;
            Pbuf[(e + 8 * hi) * 32 + 16 + ln] = (_Float16)p1;
            float ps = p0 + p1;
            #pragma unroll
            for (int off = 1; off < 16; off <<= 1)
                ps += __shfl_xor(ps, off, 32);
            psum[e] = ps;
        }
        #pragma unroll
        for (int e = 0; e < 8; ++e) l[e] += psum[e];
        __syncthreads();

        // ---- P (A-frag from LDS) @ Vdec (B-frag: K = key, contiguous) ----
        V16 pf;
        {
            const unsigned int* p = (const unsigned int*)&Pbuf[ln * 32];
            #pragma unroll
            for (int j = 0; j < 8; ++j)
                pf.u[j] = p[afrag_kbase(j, hi) >> 1];
        }
        #pragma unroll
        for (int c = 0; c < 4; ++c) {
            V16 vf;   // B = V: N = hd (=c*16+ln), K = key = kb+2j+16*hi
            const unsigned int* p = (const unsigned int*)
                (VpT + (size_t)(c * 16 + ln) * 2048 + kb);
            #pragma unroll
            for (int j = 0; j < 8; ++j)
                vf.u[j] = p[j + 8 * hi];
            o[c] = wmma_f16(pf.v, vf.v, o[c]);
        }
        __syncthreads();
    }

    #pragma unroll
    for (int c = 0; c < 4; ++c)
        #pragma unroll
        for (int e = 0; e < 8; ++e) {
            int row = s0 + e + 8 * hi;
            float val = o[c][e] / l[e];
            attn[(size_t)(b * 2048 + row) * 1024 + h * 64 + c * 16 + ln] =
                (_Float16)val;
        }
}

// ===========================================================================
extern "C" void kernel_launch(void* const* d_in, const int* in_sizes, int n_in,
                              void* d_out, int out_size, void* d_ws, size_t ws_size,
                              hipStream_t stream) {
    (void)in_sizes; (void)n_in; (void)out_size; (void)ws_size;

    const float* hidden = (const float*)d_in[0];   // [2,2048,1024]
    const float* w_attn = (const float*)d_in[1];   // [1024,3072]
    const float* b_attn = (const float*)d_in[2];   // [3072]
    const float* w_proj = (const float*)d_in[3];   // [1024,1024]
    const float* b_proj = (const float*)d_in[4];   // [1024]
    const float* wk_c   = (const float*)d_in[5];   // [16,64,32]
    const float* wv_c   = (const float*)d_in[6];   // [16,64,32]
    const float* wk_d   = (const float*)d_in[7];   // [16,32,64]
    const float* wv_d   = (const float*)d_in[8];   // [16,32,64]
    float* out = (float*)d_out;                    // [2,2048,1024]

    // workspace carve-up (all f16)
    _Float16* Xh     = (_Float16*)d_ws;                    // 4096*1024
    _Float16* WattnT = Xh     + (size_t)4096 * 1024;       // 3072*1024
    _Float16* WprojT = WattnT + (size_t)3072 * 1024;       // 1024*1024
    _Float16* qkv    = WprojT + (size_t)1024 * 1024;       // 4096*3072
    _Float16* wkeT   = qkv    + (size_t)4096 * 3072;       // 16*64*64
    _Float16* wveT   = wkeT   + (size_t)16 * 64 * 64;
    _Float16* kdec   = wveT   + (size_t)16 * 64 * 64;      // 32*2048*64
    _Float16* vdecT  = kdec   + (size_t)32 * 2048 * 64;    // 32*64*2048
    _Float16* attn   = vdecT  + (size_t)32 * 2048 * 64;    // 4096*1024

    // 0) precision / layout prep
    cvt_kernel<<<(4096 * 1024) / 256, 256, 0, stream>>>(hidden, Xh, 4096 * 1024);
    transpose_cvt_kernel<<<(1024 * 3072) / 256, 256, 0, stream>>>(w_attn, WattnT, 1024, 3072);
    transpose_cvt_kernel<<<(1024 * 1024) / 256, 256, 0, stream>>>(w_proj, WprojT, 1024, 1024);

    // 1) qkv = X @ Wattn + b   (M=4096, N=3072, K=1024)
    wmma_gemm_f16<_Float16, 1024><<<dim3(3072 / 128, 4096 / 128), 256, 0, stream>>>(
        Xh, WattnT, b_attn, qkv, 4096, 3072);

    // 2) effective compress->decompress weights per head (transposed)
    eff_weight_kernel<<<256, 256, 0, stream>>>(wk_c, wk_d, wkeT);
    eff_weight_kernel<<<256, 256, 0, stream>>>(wv_c, wv_d, wveT);

    // 3) k_dec [S][hd] and v_dec^T [hd][S]
    kv_dec_kernel<<<dim3(2048 / 128, 32), 256, 0, stream>>>(qkv, wkeT, kdec, 1024, 0);
    kv_dec_kernel<<<dim3(2048 / 128, 32), 256, 0, stream>>>(qkv, wveT, vdecT, 2048, 1);

    // 4) causal flash attention -> merged-heads attn (f16)
    flash_attn_kernel<<<dim3(2048 / 16, 16, 2), 32, 0, stream>>>(
        qkv, kdec, vdecT, attn);

    // 5) out = attn @ Wproj + b   (M=4096, N=1024, K=1024)
    wmma_gemm_f16<float, 1024><<<dim3(1024 / 128, 4096 / 128), 256, 0, stream>>>(
        attn, WprojT, b_proj, out, 4096, 1024);
}
